// DPC_25529285607736
// MI455X (gfx1250) — matrix operand order repaired and checked by reading, not compiled, not run on hardware
//
#include <hip/hip_runtime.h>
#include <hip/hip_bf16.h>
#include <math.h>

// ---------------- problem constants ----------------
#define BB   512   // batch
#define TT   64    // seq len
#define DD   784   // input dim
#define DP   800   // padded input dim (multiple of 32 for WMMA K-loop)
#define LDIM 256   // lower dim
#define HD   64    // higher dim
#define KMIX 5     // number of mixing matrices
#define H1D  128   // hyper hidden
#define NG   192   // 3*HD (GRU gates)

typedef __bf16 bf16_t;
typedef __attribute__((ext_vector_type(16))) __bf16 v16bf;
typedef __attribute__((ext_vector_type(8)))  __bf16 v8bf;
typedef __attribute__((ext_vector_type(8)))  float  v8f;

// ---------------- workspace layout (bytes, 256B-aligned) ----------------
static constexpr size_t OFF_WPRED = 0;                      // 784*256 bf16  = 401408
static constexpr size_t OFF_WIH   = 401408;                 // 192*800 bf16  = 307200
static constexpr size_t OFF_VK    = 708608;                 // 5*256*256 bf16= 655360
static constexpr size_t OFF_RTF   = 1363968;                // 2*512*256 f32 = 1048576
static constexpr size_t OFF_RTB   = 2412544;                // 2*512*256 bf16= 524288
static constexpr size_t OFF_RH    = 2936832;                // 512*64 f32    = 131072
static constexpr size_t OFF_ERR   = 3067904;                // 512*800 bf16  = 819200
static constexpr size_t OFF_GI    = 3887104;                // 512*192 f32   = 393216
static constexpr size_t OFF_WMIX  = 4280320;                // 512*8 f32     = 16384
static constexpr size_t OFF_DIG   = 4296704;                // 512*10 f32    = 20480
static constexpr size_t OFF_DIR   = 4317184;                // 512*4 f32     = 8192
static constexpr size_t OFF_ESUM  = 4325376;                // scalar accum

// ---------------- WMMA fragment loaders (wave32; contiguous k order, same for A and B) ----------------
union AFragU { v16bf v; v8bf h[2]; };

// A-matrix 16x32 bf16: lane L -> row M = L&15, half = L>>4;
// element e (0..15) at K = half*16 + e
__device__ __forceinline__ v16bf load_a_frag(const bf16_t* __restrict__ A, int lda,
                                             int m0, int kk, int lane) {
  int mr = lane & 15, half = lane >> 4;
  const bf16_t* p = A + (size_t)(m0 + mr) * lda + kk + half * 16;
  AFragU f;
  f.v = *(const v16bf*)(p);
  return f.v;
}

// B-matrix 32x16 bf16 (K x N): lane L -> col N = L&15, half = L>>4;
// element e (0..15) at K = half*16 + e  -> contiguous in W[n, k] row-major [N,K]
__device__ __forceinline__ v16bf load_b_frag(const bf16_t* __restrict__ W, int ldw,
                                             int n0, int kk, int lane) {
  int nr = lane & 15, half = lane >> 4;
  const bf16_t* p = W + (size_t)(n0 + nr) * ldw + kk + half * 16;
  return *(const v16bf*)p;
}

// ---------------- prep kernels ----------------
__global__ void k_zero_bytes(char* __restrict__ p, int n) {
  int i = blockIdx.x * blockDim.x + threadIdx.x;
  if (i < n) p[i] = 0;
}

__global__ void k_cvt(const float* __restrict__ s, bf16_t* __restrict__ d, int n) {
  int i = blockIdx.x * blockDim.x + threadIdx.x;
  if (i < n) d[i] = (bf16_t)s[i];
}

__global__ void k_cvt_pad(const float* __restrict__ s, bf16_t* __restrict__ d,
                          int rows, int cols, int stride) {
  int i = blockIdx.x * blockDim.x + threadIdx.x;
  if (i >= rows * stride) return;
  int r = i / stride, c = i % stride;
  d[i] = (c < cols) ? (bf16_t)s[(size_t)r * cols + c] : (bf16_t)0.f;
}

// ---------------- step 1: pred = rt @ W_pred^T ; error = x_t - pred ----------------
// grid (49, 32), block 32 (one wave, EXEC all-1s for WMMA)
__global__ __launch_bounds__(32)
void k_step1(const float* __restrict__ x_seq, int t,
             const bf16_t* __restrict__ rt_bf, const bf16_t* __restrict__ Wp,
             bf16_t* __restrict__ err_bf, float* __restrict__ errsum) {
  int lane = threadIdx.x;
  int n0 = blockIdx.x * 16;   // D tile
  int m0 = blockIdx.y * 16;   // batch tile
  v8f acc = {};
  #pragma unroll
  for (int kk = 0; kk < LDIM; kk += 32) {
    v16bf a = load_a_frag(rt_bf, LDIM, m0, kk, lane);
    v16bf b = load_b_frag(Wp, LDIM, n0, kk, lane);
    acc = __builtin_amdgcn_wmma_f32_16x16x32_bf16(false, a, false, b, (short)0, acc, false, false);
  }
  int nr = lane & 15, half = lane >> 4;
  float lsum = 0.f;
  #pragma unroll
  for (int r = 0; r < 8; ++r) {
    int m = m0 + half * 8 + r;
    int n = n0 + nr;
    // x_seq is [B, T, D]
    float e = x_seq[((size_t)m * TT + t) * DD + n] - acc[r];
    err_bf[(size_t)m * DP + n] = (bf16_t)e;
    lsum += e * e;
  }
  atomicAdd(errsum, lsum);
}

// ---------------- step 2a: gi = error @ W_ih^T (K padded to 800) ----------------
// grid (12, 32), block 32
__global__ __launch_bounds__(32)
void k_step2a(const bf16_t* __restrict__ err_bf, const bf16_t* __restrict__ Wih,
              float* __restrict__ gi) {
  int lane = threadIdx.x;
  int n0 = blockIdx.x * 16;
  int m0 = blockIdx.y * 16;
  v8f acc = {};
  for (int kk = 0; kk < DP; kk += 32) {
    v16bf a = load_a_frag(err_bf, DP, m0, kk, lane);
    v16bf b = load_b_frag(Wih, DP, n0, kk, lane);
    acc = __builtin_amdgcn_wmma_f32_16x16x32_bf16(false, a, false, b, (short)0, acc, false, false);
  }
  int nr = lane & 15, half = lane >> 4;
  #pragma unroll
  for (int r = 0; r < 8; ++r) {
    int m = m0 + half * 8 + r;
    gi[(size_t)m * NG + n0 + nr] = acc[r];
  }
}

// ---------------- step 2b: GRU gates + hyper-network + direction head ----------------
// grid 512 (one block per batch row), block 64
__global__ __launch_bounds__(64)
void k_step2b(const float* __restrict__ gi, float* __restrict__ rh,
              const float* __restrict__ Whh, const float* __restrict__ bih,
              const float* __restrict__ bhh,
              const float* __restrict__ W1, const float* __restrict__ b1,
              const float* __restrict__ W2, const float* __restrict__ b2,
              const float* __restrict__ Wr, const float* __restrict__ br,
              float* __restrict__ wmix, float* __restrict__ dir_acc) {
  int b = blockIdx.x, tid = threadIdx.x;
  __shared__ float s_rh[HD], s_h[HD], s_h1[H1D];
  s_rh[tid] = rh[(size_t)b * HD + tid];
  __syncthreads();

  float ir  = gi[(size_t)b * NG + tid]          + bih[tid];
  float iz  = gi[(size_t)b * NG + HD + tid]     + bih[HD + tid];
  float inn = gi[(size_t)b * NG + 2 * HD + tid] + bih[2 * HD + tid];
  float hr = bhh[tid], hz = bhh[HD + tid], hn = bhh[2 * HD + tid];
  for (int j = 0; j < HD; ++j) {
    float rj = s_rh[j];
    hr += Whh[(size_t)tid * HD + j] * rj;
    hz += Whh[(size_t)(HD + tid) * HD + j] * rj;
    hn += Whh[(size_t)(2 * HD + tid) * HD + j] * rj;
  }
  float r = 1.f / (1.f + __expf(-(ir + hr)));
  float z = 1.f / (1.f + __expf(-(iz + hz)));
  float n = tanhf(inn + r * hn);
  float hnew = (1.f - z) * n + z * s_rh[tid];
  s_h[tid] = hnew;
  rh[(size_t)b * HD + tid] = hnew;
  __syncthreads();

  if (tid < 4) {   // direction logits accumulation
    float s = br[tid];
    for (int j = 0; j < HD; ++j) s += Wr[tid * HD + j] * s_h[j];
    dir_acc[b * 4 + tid] += s;
  }
  for (int o = tid; o < H1D; o += 64) {   // hyper layer 1 + ReLU
    float s = b1[o];
    for (int j = 0; j < HD; ++j) s += W1[o * HD + j] * s_h[j];
    s_h1[o] = fmaxf(s, 0.f);
  }
  __syncthreads();
  if (tid < KMIX) {  // mixing weights
    float s = b2[tid];
    for (int j = 0; j < H1D; ++j) s += W2[tid * H1D + j] * s_h1[j];
    wmix[b * 8 + tid] = s;
  }
}

// ---------------- step 3: mixed = sum_k w_k * (rt @ Vk^T); rt_new = relu + noise ----------------
// grid (16, 32), block 32; five f32 accumulators share each A fragment
__global__ __launch_bounds__(32)
void k_step3(const bf16_t* __restrict__ rt_bf_cur, const bf16_t* __restrict__ Vkb,
             const float* __restrict__ wmix, const float* __restrict__ noise_t,
             float* __restrict__ rt_f_nxt, bf16_t* __restrict__ rt_bf_nxt) {
  int lane = threadIdx.x;
  int n0 = blockIdx.x * 16;
  int m0 = blockIdx.y * 16;
  v8f acc[KMIX];
  #pragma unroll
  for (int k5 = 0; k5 < KMIX; ++k5) acc[k5] = (v8f){};

  for (int kk = 0; kk < LDIM; kk += 32) {
    v16bf a = load_a_frag(rt_bf_cur, LDIM, m0, kk, lane);
    #pragma unroll
    for (int k5 = 0; k5 < KMIX; ++k5) {
      v16bf b = load_b_frag(Vkb + (size_t)k5 * LDIM * LDIM, LDIM, n0, kk, lane);
      acc[k5] = __builtin_amdgcn_wmma_f32_16x16x32_bf16(false, a, false, b, (short)0,
                                                        acc[k5], false, false);
    }
  }
  int nr = lane & 15, half = lane >> 4;
  #pragma unroll
  for (int r = 0; r < 8; ++r) {
    int m = m0 + half * 8 + r;
    int n = n0 + nr;
    float s = 0.f;
    #pragma unroll
    for (int k5 = 0; k5 < KMIX; ++k5) s += wmix[m * 8 + k5] * acc[k5][r];
    float v = fmaxf(s, 0.f) + 0.01f * noise_t[(size_t)m * LDIM + n];
    rt_f_nxt[(size_t)m * LDIM + n] = v;
    rt_bf_nxt[(size_t)m * LDIM + n] = (bf16_t)v;
  }
}

// ---------------- step 4: digit head accumulation ----------------
// grid 512, block 64
__global__ __launch_bounds__(64)
void k_step4(const float* __restrict__ rt_f, const float* __restrict__ Wd,
             const float* __restrict__ bd, float* __restrict__ dig_acc) {
  int b = blockIdx.x, tid = threadIdx.x;
  __shared__ float s_rt[LDIM];
  for (int i = tid; i < LDIM; i += 64) s_rt[i] = rt_f[(size_t)b * LDIM + i];
  __syncthreads();
  if (tid < 10) {
    float s = bd[tid];
    for (int j = 0; j < LDIM; ++j) s += Wd[tid * LDIM + j] * s_rt[j];
    dig_acc[b * 10 + tid] += s;
  }
}

// ---------------- finalize: means -> d_out ----------------
__global__ void k_final(const float* __restrict__ dig_acc, const float* __restrict__ dir_acc,
                        const float* __restrict__ errsum, float* __restrict__ out) {
  int i = blockIdx.x * blockDim.x + threadIdx.x;
  const int NDIG = BB * 10, NDIR = BB * 4;
  if (i < NDIG)                 out[i] = dig_acc[i] * (1.f / TT);
  else if (i < NDIG + NDIR)     out[i] = dir_acc[i - NDIG] * (1.f / TT);
  else if (i == NDIG + NDIR)    out[i] = errsum[0] * (1.f / ((float)TT * BB * DD));
}

// ---------------- launch ----------------
extern "C" void kernel_launch(void* const* d_in, const int* in_sizes, int n_in,
                              void* d_out, int out_size, void* d_ws, size_t ws_size,
                              hipStream_t stream) {
  const float* x_seq  = (const float*)d_in[0];
  const float* noise  = (const float*)d_in[1];
  const float* W_pred = (const float*)d_in[2];
  const float* Vk     = (const float*)d_in[3];
  const float* W1     = (const float*)d_in[4];
  const float* b1     = (const float*)d_in[5];
  const float* W2     = (const float*)d_in[6];
  const float* b2     = (const float*)d_in[7];
  const float* W_ih   = (const float*)d_in[8];
  const float* W_hh   = (const float*)d_in[9];
  const float* b_ih   = (const float*)d_in[10];
  const float* b_hh   = (const float*)d_in[11];
  const float* Wd     = (const float*)d_in[12];
  const float* bd     = (const float*)d_in[13];
  const float* Wr     = (const float*)d_in[14];
  const float* br     = (const float*)d_in[15];

  char* ws = (char*)d_ws;
  bf16_t* Wp_bf  = (bf16_t*)(ws + OFF_WPRED);
  bf16_t* Wih_bf = (bf16_t*)(ws + OFF_WIH);
  bf16_t* Vk_bf  = (bf16_t*)(ws + OFF_VK);
  float*  Rt_f   = (float*)(ws + OFF_RTF);
  bf16_t* Rt_bf  = (bf16_t*)(ws + OFF_RTB);
  float*  Rh     = (float*)(ws + OFF_RH);
  bf16_t* Err_bf = (bf16_t*)(ws + OFF_ERR);
  float*  Gi     = (float*)(ws + OFF_GI);
  float*  Wmix   = (float*)(ws + OFF_WMIX);
  float*  DigAcc = (float*)(ws + OFF_DIG);
  float*  DirAcc = (float*)(ws + OFF_DIR);
  float*  Esum   = (float*)(ws + OFF_ESUM);

  auto blocks = [](int n) { return (n + 255) / 256; };

  // --- prep: bf16 weight conversions + state/accumulator zeroing (every call; deterministic)
  k_cvt<<<blocks(DD * LDIM), 256, 0, stream>>>(W_pred, Wp_bf, DD * LDIM);
  k_cvt_pad<<<blocks(NG * DP), 256, 0, stream>>>(W_ih, Wih_bf, NG, DD, DP);
  k_cvt<<<blocks(KMIX * LDIM * LDIM), 256, 0, stream>>>(Vk, Vk_bf, KMIX * LDIM * LDIM);
  k_zero_bytes<<<blocks(BB * LDIM * 4), 256, 0, stream>>>((char*)Rt_f, BB * LDIM * 4);
  k_zero_bytes<<<blocks(BB * LDIM * 2), 256, 0, stream>>>((char*)Rt_bf, BB * LDIM * 2);
  k_zero_bytes<<<blocks(BB * HD * 4), 256, 0, stream>>>((char*)Rh, BB * HD * 4);
  k_zero_bytes<<<blocks(BB * DP * 2), 256, 0, stream>>>((char*)Err_bf, BB * DP * 2);
  k_zero_bytes<<<blocks(BB * 10 * 4), 256, 0, stream>>>((char*)DigAcc, BB * 10 * 4);
  k_zero_bytes<<<blocks(BB * 4 * 4), 256, 0, stream>>>((char*)DirAcc, BB * 4 * 4);
  k_zero_bytes<<<1, 256, 0, stream>>>((char*)Esum, 4);

  // --- recurrence over T steps (ping-pong rt buffers) ---
  for (int t = 0; t < TT; ++t) {
    int cur = t & 1, nxt = cur ^ 1;
    const bf16_t* rtb_c = Rt_bf + (size_t)cur * BB * LDIM;
    bf16_t*       rtb_n = Rt_bf + (size_t)nxt * BB * LDIM;
    float*        rtf_n = Rt_f  + (size_t)nxt * BB * LDIM;
    const float*  noi_t = noise + (size_t)t * BB * LDIM;

    k_step1<<<dim3(DD / 16, BB / 16), 32, 0, stream>>>(x_seq, t, rtb_c, Wp_bf, Err_bf, Esum);
    k_step2a<<<dim3(NG / 16, BB / 16), 32, 0, stream>>>(Err_bf, Wih_bf, Gi);
    k_step2b<<<BB, 64, 0, stream>>>(Gi, Rh, W_hh, b_ih, b_hh, W1, b1, W2, b2, Wr, br,
                                    Wmix, DirAcc);
    k_step3<<<dim3(LDIM / 16, BB / 16), 32, 0, stream>>>(rtb_c, Vk_bf, Wmix, noi_t,
                                                         rtf_n, rtb_n);
    k_step4<<<BB, 64, 0, stream>>>(rtf_n, Wd, bd, DigAcc);
  }

  k_final<<<blocks(BB * 10 + BB * 4 + 1), 256, 0, stream>>>(DigAcc, DirAcc, Esum, (float*)d_out);
}
